// CausalSelfAttention_46780783788600
// MI455X (gfx1250) — matrix operand order repaired and checked by reading, not compiled, not run on hardware
//
#include <hip/hip_runtime.h>
#include <hip/hip_bf16.h>

typedef __bf16 bf16;
typedef __attribute__((ext_vector_type(16))) __bf16 v16bf;
typedef __attribute__((ext_vector_type(8)))  __bf16 v8bf;
typedef __attribute__((ext_vector_type(8)))  float  v8f;
typedef __attribute__((ext_vector_type(4)))  int    v4i;

// gfx1250 async global->LDS copy support (ASYNCcnt path). Guarded so the code
// still compiles (direct load+ds_store fallback) if the builtins are absent.
#if defined(__AMDGCN__) && \
    __has_builtin(__builtin_amdgcn_global_load_async_to_lds_b128) && \
    __has_builtin(__builtin_amdgcn_s_wait_asynccnt)
#define USE_ASYNC_LDS 1
#else
#define USE_ASYNC_LDS 0
#endif

// Builtin signature is typed over the 16-byte payload:
//   void (*)(v4i addrspace(1)* src, v4i addrspace(3)* dst, imm offset, imm cpol)
typedef __attribute__((address_space(1))) v4i gv4i;
typedef __attribute__((address_space(3))) v4i lv4i;

// Copy 16 bytes global -> LDS.
__device__ __forceinline__ void cp16_g2l(const bf16* g, bf16* l) {
#if USE_ASYNC_LDS
  __builtin_amdgcn_global_load_async_to_lds_b128((gv4i*)g, (lv4i*)l, 0, 0);
#else
  *(float4*)l = *(const float4*)g;
#endif
}

__device__ __forceinline__ void async_fence() {
#if USE_ASYNC_LDS
  __builtin_amdgcn_s_wait_asynccnt(0);
#endif
}

// ---------------------------------------------------------------------------
// float -> bf16 (round-to-nearest-even), bit-level so we only need __bf16 as
// a storage type.
// ---------------------------------------------------------------------------
__device__ __forceinline__ bf16 f2bf(float f) {
  union { float f; unsigned u; } in; in.f = f;
  unsigned u = in.u;
  unsigned r = u + 0x7FFFu + ((u >> 16) & 1u);
  union { unsigned short s; bf16 b; } out;
  out.s = (unsigned short)(r >> 16);
  return out.b;
}

// ---------------------------------------------------------------------------
// Elementwise f32 -> bf16
// ---------------------------------------------------------------------------
__global__ void cvt_bf16_kernel(const float* __restrict__ in,
                                bf16* __restrict__ out, int n) {
  int i = blockIdx.x * 256 + threadIdx.x;
  if (i < n) out[i] = f2bf(in[i]);
}

// ---------------------------------------------------------------------------
// W[k][n] (f32, KxN row-major) -> Wt[n][k] (bf16) so B-matrix fragments are
// contiguous per-lane loads.
// ---------------------------------------------------------------------------
__global__ void transpose_cvt_kernel(const float* __restrict__ w,
                                     bf16* __restrict__ wt, int K, int N) {
  int idx = blockIdx.x * 256 + threadIdx.x;
  if (idx >= K * N) return;
  int n = idx / K;
  int k = idx - n * K;
  wt[idx] = f2bf(w[(size_t)k * N + n]);
}

// ---------------------------------------------------------------------------
// Generic bf16 GEMM, f32 accumulate:  C[M][N] = A[M][K] * Bt[N][K]^T
// Block tile 128x128, K-tile 32, 256 threads = 8 wave32.
// Each wave: 32x64 tile = 2x4 v_wmma_f32_16x16x32_bf16 per K-step.
// Tiles staged through LDS via global_load_async_to_lds_b128 (ASYNCcnt).
// ---------------------------------------------------------------------------
__global__ __launch_bounds__(256)
void gemm_bf16_kernel(const bf16* __restrict__ A, const bf16* __restrict__ Bt,
                      float* __restrict__ C, int M, int N, int K) {
  __shared__ bf16 As[128 * 32];
  __shared__ bf16 Bs[128 * 32];

  const int tid  = threadIdx.x;
  const int lane = tid & 31;
  const int w    = tid >> 5;
  const int r    = lane & 15;     // row/col within a 16-tile
  const int hsel = lane >> 4;     // which half-wave

  const int m0 = blockIdx.y * 128;
  const int n0 = blockIdx.x * 128;
  const int mw = (w & 3) * 32;    // wave's m offset in block
  const int nw = (w >> 2) * 64;   // wave's n offset in block

  // Static staging assignment: thread covers rows (tid>>2) and (tid>>2)+64,
  // 16B chunk (tid&3)*8 elements into the 32-element (64 B) LDS row.
  const int srow = tid >> 2;
  const int soff = (tid & 3) * 8;

  v8f acc[2][4];
#pragma unroll
  for (int i = 0; i < 2; i++)
#pragma unroll
    for (int j = 0; j < 4; j++)
#pragma unroll
      for (int e = 0; e < 8; e++) acc[i][j][e] = 0.0f;

  for (int kt = 0; kt < K; kt += 32) {
    // Stage A tile [128 rows][32 k] and Bt tile [128 n][32 k] into LDS.
    cp16_g2l(&A[(size_t)(m0 + srow) * K + kt + soff], &As[srow * 32 + soff]);
    cp16_g2l(&A[(size_t)(m0 + srow + 64) * K + kt + soff],
             &As[(srow + 64) * 32 + soff]);
    cp16_g2l(&Bt[(size_t)(n0 + srow) * K + kt + soff], &Bs[srow * 32 + soff]);
    cp16_g2l(&Bt[(size_t)(n0 + srow + 64) * K + kt + soff],
             &Bs[(srow + 64) * 32 + soff]);
    if (kt + 32 < K) {
      __builtin_prefetch(&A[(size_t)(m0 + srow) * K + kt + 32 + soff], 0, 0);
      __builtin_prefetch(&Bt[(size_t)(n0 + srow) * K + kt + 32 + soff], 0, 0);
    }
    async_fence();
    __syncthreads();

    // A fragments: lane holds row (mw+i*16+r); contiguous 16 k at hsel*16.
    v16bf af[2];
#pragma unroll
    for (int i = 0; i < 2; i++) {
      const bf16* p = &As[(mw + i * 16 + r) * 32];
      v8bf lo = *(const v8bf*)(p + hsel * 16);
      v8bf hi = *(const v8bf*)(p + hsel * 16 + 8);
#pragma unroll
      for (int e = 0; e < 8; e++) { af[i][e] = lo[e]; af[i][e + 8] = hi[e]; }
    }
    // B fragments: lane holds col (nw+j*16+r); contiguous 16 k at hsel*16.
    v16bf bfg[4];
#pragma unroll
    for (int j = 0; j < 4; j++)
      bfg[j] = *(const v16bf*)(&Bs[(nw + j * 16 + r) * 32 + hsel * 16]);

#pragma unroll
    for (int i = 0; i < 2; i++)
#pragma unroll
      for (int j = 0; j < 4; j++)
        acc[i][j] = __builtin_amdgcn_wmma_f32_16x16x32_bf16(
            false, af[i], false, bfg[j], (short)0, acc[i][j], false, false);
    __syncthreads();
  }

  // C layout: vgpr p, lane -> row = p + 8*hsel, col = r.
#pragma unroll
  for (int i = 0; i < 2; i++)
#pragma unroll
    for (int j = 0; j < 4; j++)
#pragma unroll
      for (int p = 0; p < 8; p++)
        C[(size_t)(m0 + mw + i * 16 + p + 8 * hsel) * N +
          (n0 + nw + j * 16 + r)] = acc[i][j][p];
}

// ---------------------------------------------------------------------------
// RoPE + RMSNorm for one (b, t, head) row of 128, one wave per row.
// in : [b][t][NH][128] f32   out: [b][NH][t][128] bf16
// ---------------------------------------------------------------------------
__global__ void rope_rms_kernel(const float* __restrict__ in,
                                const float* __restrict__ cs,
                                const float* __restrict__ sn,
                                bf16* __restrict__ out, int T, int NH) {
  int warp = (blockIdx.x * blockDim.x + threadIdx.x) >> 5;
  int lane = threadIdx.x & 31;
  int h = warp % NH;
  int t = (warp / NH) % T;
  int b = warp / (NH * T);

  const float* row = in + (size_t)warp * 128;
  int i0 = lane, i1 = lane + 32;

  float o0, o1, o2, o3;
  {
    float t1 = row[i0], t2 = row[i0 + 64];
    float c = cs[t * 64 + i0], s = sn[t * 64 + i0];
    o0 = t1 * c + t2 * s;  o2 = -t1 * s + t2 * c;
  }
  {
    float t1 = row[i1], t2 = row[i1 + 64];
    float c = cs[t * 64 + i1], s = sn[t * 64 + i1];
    o1 = t1 * c + t2 * s;  o3 = -t1 * s + t2 * c;
  }
  float ss = o0 * o0 + o1 * o1 + o2 * o2 + o3 * o3;
#pragma unroll
  for (int off = 16; off >= 1; off >>= 1) ss += __shfl_xor(ss, off, 32);
  float rn = rsqrtf(ss * (1.0f / 128.0f) + 1.1920929e-07f);

  bf16* orow = out + (((size_t)(b * NH + h)) * T + t) * 128;
  orow[i0]      = f2bf(o0 * rn);
  orow[i1]      = f2bf(o1 * rn);
  orow[i0 + 64] = f2bf(o2 * rn);
  orow[i1 + 64] = f2bf(o3 * rn);
}

// ---------------------------------------------------------------------------
// V: [b][t][HK][128] f32  ->  [b][HK][128][t] bf16 (transposed so PV B-frags
// are contiguous along t)
// ---------------------------------------------------------------------------
__global__ void v_transform_kernel(const float* __restrict__ vf,
                                   bf16* __restrict__ vt, int T, int HK) {
  int idx = blockIdx.x * 256 + threadIdx.x;
  int d  = idx & 127;
  int hk = (idx >> 7) % HK;
  int t  = (idx >> 7) / HK % T;
  int b  = idx / (128 * HK * T);
  vt[(((size_t)(b * HK + hk)) * 128 + d) * T + t] = f2bf(vf[idx]);
}

// ---------------------------------------------------------------------------
// Flash attention, causal, GQA. One wave per 16 query rows.
// q: [b][H][t][128] bf16, k: [b][HK][t][128] bf16, v: [b][HK][128][t] bf16
// y: [b][t][H*128] bf16
// ---------------------------------------------------------------------------
__global__ __launch_bounds__(32)
void attn_kernel(const bf16* __restrict__ q, const bf16* __restrict__ k,
                 const bf16* __restrict__ v, bf16* __restrict__ y,
                 int T, int H, int HK) {
  __shared__ bf16 Pl[16 * 32];

  const int lane = threadIdx.x;
  const int r    = lane & 15;
  const int hsel = lane >> 4;
  const int qt = blockIdx.x, h = blockIdx.y, b = blockIdx.z;
  const int hk = h / (H / HK);
  const int m0 = qt * 16;
  const float scale = 0.08838834764831845f;  // 1/sqrt(128)
  const float NEG_INF = -__builtin_inff();

  // Q A-fragments (4 chunks of 32 over D=128), straight from global.
  const bf16* qrow = q + (((size_t)(b * H + h)) * T + m0 + r) * 128;
  v16bf qa[4];
#pragma unroll
  for (int kc = 0; kc < 4; kc++) {
    v8bf lo = *(const v8bf*)(qrow + kc * 32 + hsel * 16);
    v8bf hi = *(const v8bf*)(qrow + kc * 32 + hsel * 16 + 8);
#pragma unroll
    for (int e = 0; e < 8; e++) { qa[kc][e] = lo[e]; qa[kc][e + 8] = hi[e]; }
  }

  v8f o[8];
  float mi[8], li[8];
#pragma unroll
  for (int dt = 0; dt < 8; dt++)
#pragma unroll
    for (int e = 0; e < 8; e++) o[dt][e] = 0.0f;
#pragma unroll
  for (int p = 0; p < 8; p++) { mi[p] = NEG_INF; li[p] = 0.0f; }

  const bf16* kbase = k + ((size_t)(b * HK + hk)) * T * 128;
  const bf16* vbase = v + ((size_t)(b * HK + hk)) * 128 * T;

  const int nb = (m0 + 15) / 32 + 1;  // causal: only blocks intersecting tril
  for (int j = 0; j < nb; j++) {
    const int kb0 = j * 32;

    // S = Q K^T : two 16-col tiles, contraction D=128 in 4 WMMAs each.
    v8f s[2];
#pragma unroll
    for (int nt = 0; nt < 2; nt++) {
#pragma unroll
      for (int e = 0; e < 8; e++) s[nt][e] = 0.0f;
#pragma unroll
      for (int kc = 0; kc < 4; kc++) {
        v16bf kf = *(const v16bf*)(kbase + (size_t)(kb0 + nt * 16 + r) * 128 +
                                   kc * 32 + hsel * 16);
        s[nt] = __builtin_amdgcn_wmma_f32_16x16x32_bf16(
            false, qa[kc], false, kf, (short)0, s[nt], false, false);
      }
    }

    const bool need_mask = (kb0 + 31 > m0);
    float corr[8];
#pragma unroll
    for (int p = 0; p < 8; p++) {
      int row = m0 + p + 8 * hsel;
      float x0 = s[0][p] * scale;
      float x1 = s[1][p] * scale;
      if (need_mask) {
        if (kb0 + r > row)      x0 = NEG_INF;
        if (kb0 + 16 + r > row) x1 = NEG_INF;
      }
      float bx = fmaxf(x0, x1);
#pragma unroll
      for (int off = 8; off >= 1; off >>= 1)
        bx = fmaxf(bx, __shfl_xor(bx, off, 16));
      float mnew = fmaxf(mi[p], bx);
      corr[p] = __expf(mi[p] - mnew);
      float p0 = __expf(x0 - mnew);
      float p1 = __expf(x1 - mnew);
      float rs = p0 + p1;
#pragma unroll
      for (int off = 8; off >= 1; off >>= 1) rs += __shfl_xor(rs, off, 16);
      li[p] = li[p] * corr[p] + rs;
      mi[p] = mnew;
      // stash P (C-layout -> LDS tile [16 rows][32 cols], bf16)
      Pl[(p + 8 * hsel) * 32 + r]      = f2bf(p0);
      Pl[(p + 8 * hsel) * 32 + 16 + r] = f2bf(p1);
    }
#pragma unroll
    for (int dt = 0; dt < 8; dt++)
#pragma unroll
      for (int p = 0; p < 8; p++) o[dt][p] *= corr[p];

    __syncthreads();  // single-wave WG: orders DS store -> DS load

    // reload P as an A fragment (16x32)
    v16bf pa;
    {
      const bf16* pr = &Pl[r * 32];
      v8bf lo = *(const v8bf*)(pr + hsel * 16);
      v8bf hi = *(const v8bf*)(pr + hsel * 16 + 8);
#pragma unroll
      for (int e = 0; e < 8; e++) { pa[e] = lo[e]; pa[e + 8] = hi[e]; }
    }

    // O += P V : 8 d-tiles, contraction over 32 keys.
#pragma unroll
    for (int dt = 0; dt < 8; dt++) {
      v16bf vf16 = *(const v16bf*)(vbase + (size_t)(dt * 16 + r) * T + kb0 +
                                   hsel * 16);
      o[dt] = __builtin_amdgcn_wmma_f32_16x16x32_bf16(
          false, pa, false, vf16, (short)0, o[dt], false, false);
    }
    __syncthreads();
  }

  // epilogue: divide by row sum, write y[b][t][h*128+d] bf16
#pragma unroll
  for (int dt = 0; dt < 8; dt++)
#pragma unroll
    for (int p = 0; p < 8; p++) {
      int row = m0 + p + 8 * hsel;
      y[((size_t)(b * T + row)) * (size_t)(H * 128) + h * 128 + dt * 16 + r] =
          f2bf(o[dt][p] * (1.0f / li[p]));
    }
}

// ---------------------------------------------------------------------------
// Host orchestration
// ---------------------------------------------------------------------------
extern "C" void kernel_launch(void* const* d_in, const int* in_sizes, int n_in,
                              void* d_out, int out_size, void* d_ws,
                              size_t ws_size, hipStream_t stream) {
  (void)in_sizes; (void)n_in; (void)out_size; (void)ws_size;
  const float* x    = (const float*)d_in[0];
  const float* cosb = (const float*)d_in[1];
  const float* sinb = (const float*)d_in[2];
  const float* wq   = (const float*)d_in[3];
  const float* wk   = (const float*)d_in[4];
  const float* wv   = (const float*)d_in[5];
  const float* wo   = (const float*)d_in[6];
  float* out = (float*)d_out;

  constexpr int B = 2, T = 2048, C = 2048, H = 16, HK = 8, D = 128;
  constexpr int M  = B * T;       // 4096
  constexpr int NQ = H * D;       // 2048
  constexpr int NK = HK * D;      // 1024

  char* ws = (char*)d_ws;
  size_t off = 0;
  auto alloc = [&](size_t bytes) -> void* {
    void* p = ws + off;
    off += (bytes + 255) & ~(size_t)255;
    return p;
  };

  bf16*  xb  = (bf16*)alloc((size_t)M * C * 2);
  bf16*  wqt = (bf16*)alloc((size_t)NQ * C * 2);
  bf16*  wkt = (bf16*)alloc((size_t)NK * C * 2);
  bf16*  wvt = (bf16*)alloc((size_t)NK * C * 2);
  bf16*  wot = (bf16*)alloc((size_t)C * C * 2);
  float* qf  = (float*)alloc((size_t)M * NQ * 4);
  float* kf  = (float*)alloc((size_t)M * NK * 4);
  float* vf  = (float*)alloc((size_t)M * NK * 4);
  bf16*  qn  = (bf16*)alloc((size_t)B * H * T * D * 2);
  bf16*  kn  = (bf16*)alloc((size_t)B * HK * T * D * 2);
  bf16*  vt  = (bf16*)alloc((size_t)B * HK * D * T * 2);
  bf16*  yb  = (bf16*)qf;  // qf is dead after rope_rms(q); reuse its space

  // 1) precision conversion + weight transposes
  cvt_bf16_kernel<<<(M * C) / 256, 256, 0, stream>>>(x, xb, M * C);
  transpose_cvt_kernel<<<(C * NQ) / 256, 256, 0, stream>>>(wq, wqt, C, NQ);
  transpose_cvt_kernel<<<(C * NK) / 256, 256, 0, stream>>>(wk, wkt, C, NK);
  transpose_cvt_kernel<<<(C * NK) / 256, 256, 0, stream>>>(wv, wvt, C, NK);
  transpose_cvt_kernel<<<(C * C) / 256, 256, 0, stream>>>(wo, wot, C, C);

  // 2) QKV projections (bf16 WMMA, f32 accumulate)
  gemm_bf16_kernel<<<dim3(NQ / 128, M / 128), 256, 0, stream>>>(xb, wqt, qf, M, NQ, C);
  gemm_bf16_kernel<<<dim3(NK / 128, M / 128), 256, 0, stream>>>(xb, wkt, kf, M, NK, C);
  gemm_bf16_kernel<<<dim3(NK / 128, M / 128), 256, 0, stream>>>(xb, wvt, vf, M, NK, C);

  // 3) RoPE + RMSNorm (+ head-major relayout), V transpose
  rope_rms_kernel<<<(B * T * H) / 8, 256, 0, stream>>>(qf, cosb, sinb, qn, T, H);
  rope_rms_kernel<<<(B * T * HK) / 8, 256, 0, stream>>>(kf, cosb, sinb, kn, T, HK);
  v_transform_kernel<<<(B * T * HK * D) / 256, 256, 0, stream>>>(vf, vt, T, HK);

  // 4) causal flash attention (WMMA QK^T and PV)
  attn_kernel<<<dim3(T / 16, H, B), 32, 0, stream>>>(qn, kn, vt, yb, T, H, HK);

  // 5) output projection
  gemm_bf16_kernel<<<dim3(C / 128, M / 128), 256, 0, stream>>>(yb, wot, out, M, C, C);
}
